// torch_sigma_86088324481370
// MI455X (gfx1250) — matrix operand order repaired and checked by reading, not compiled, not run on hardware
//
#include <hip/hip_runtime.h>
#include <hip/hip_bf16.h>

// ---------------------------------------------------------------------------
// CDNA5 (gfx1250) implementation. wave32. WMMA f32<-f16 16x16x32 for all GEMMs.
// All WMMA B-operands are loaded contiguously (producers emit transposed f16
// copies where needed), so every fragment load is 2x global_load_b128.
// ---------------------------------------------------------------------------

typedef __attribute__((ext_vector_type(16))) _Float16 v16h;
typedef __attribute__((ext_vector_type(8)))  float    v8f;

#define DEVINL __device__ __forceinline__

// A fragment: row-major [.. x ld], tile origin p=(m0,k0), 16x32 f16.
// ISA layout: lanes 0-15 row m=lane, K in {0..7}U{16..23}; lanes 16-31 row
// m=lane-16, K in {8..15}U{24..31}.
DEVINL v16h load_fragA(const _Float16* __restrict__ p, int ld, int lane) {
  int m = lane & 15;
  int koff = (lane >> 4) << 3;
  const _Float16* r = p + (size_t)m * ld + koff;
  v16h out;
#pragma unroll
  for (int j = 0; j < 8; ++j) { out[j] = r[j]; out[8 + j] = r[16 + j]; }
  return out;
}

// B fragment where B[k][n] = W[n][k], W row-major [N x K], tile origin w=(n0,k0).
// ISA layout: lane holds column n=lane&15; lanes 0-15 K in {0..7}U{16..23}, lanes 16-31 K in {8..15}U{24..31}.
// 2x 16 contiguous bytes per lane -> 2x b128.
DEVINL v16h load_fragB_W(const _Float16* __restrict__ w, int ld, int lane) {
  int n = lane & 15;
  int kb = (lane >> 4) << 3;
  const _Float16* r = w + (size_t)n * ld + kb;
  v16h out;
#pragma unroll
  for (int j = 0; j < 8; ++j) { out[j] = r[j]; out[8 + j] = r[16 + j]; }
  return out;
}

DEVINL v8f wmma_f16(v16h a, v16h b, v8f c) {
  return __builtin_amdgcn_wmma_f32_16x16x32_f16(false, a, false, b, (short)0, c,
                                                false, false);
}

// ---------------------------------------------------------------------------
// Conv 3x3 (VALID) + ReLU + 2x2/2 maxpool, fp32. grid=(1024, Cout), block=256.
// dyn-LDS = Cin*9 floats (weights of this out-channel).
// ---------------------------------------------------------------------------
__global__ void conv_relu_pool(const float* __restrict__ in, const float* __restrict__ w,
                               const float* __restrict__ bias, float* __restrict__ out,
                               int Cin, int Sin) {
  int img = blockIdx.x;
  int co  = blockIdx.y;
  int Cout = gridDim.y;
  int Sc = Sin - 2, Sp = Sc >> 1;
  extern __shared__ float wsm[];
  for (int t = threadIdx.x; t < Cin * 9; t += blockDim.x)
    wsm[t] = w[(size_t)co * Cin * 9 + t];
  __syncthreads();
  float bb = bias[co];
  const float* ib = in + (size_t)img * Cin * Sin * Sin;
  float* ob = out + ((size_t)img * Cout + co) * Sp * Sp;
  for (int p = threadIdx.x; p < Sp * Sp; p += blockDim.x) {
    int py = p / Sp, px = p % Sp;
    float mx = -1e30f;
#pragma unroll
    for (int dy = 0; dy < 2; ++dy)
#pragma unroll
      for (int dx = 0; dx < 2; ++dx) {
        int oy = 2 * py + dy, ox = 2 * px + dx;
        float v = bb;
        for (int ci = 0; ci < Cin; ++ci) {
          const float* ip = ib + ((size_t)ci * Sin + oy) * Sin + ox;
          const float* wp = wsm + ci * 9;
          v += ip[0] * wp[0] + ip[1] * wp[1] + ip[2] * wp[2]
             + ip[Sin] * wp[3] + ip[Sin + 1] * wp[4] + ip[Sin + 2] * wp[5]
             + ip[2 * Sin] * wp[6] + ip[2 * Sin + 1] * wp[7] + ip[2 * Sin + 2] * wp[8];
        }
        mx = fmaxf(mx, v);
      }
    ob[p] = fmaxf(mx, 0.f);  // relu(max) == max(relu) (monotone)
  }
}

// conv4 (10ch, 10x10 -> 8x8 -> pool 4x4), packed into Hf[img][160] (f32 + f16)
// plus per-batch transposed f16 copy HfT[b][feat][node]. grid=1024, block=160.
__global__ void conv4_pack(const float* __restrict__ in, const float* __restrict__ w,
                           const float* __restrict__ bias,
                           float* __restrict__ Hf, _Float16* __restrict__ Hfh,
                           _Float16* __restrict__ HfT) {
  const int Sin = 10, Cin = 10;
  int img = blockIdx.x;
  int t = threadIdx.x;           // 0..159 : feature = co*16 + py*4 + px
  int co = t >> 4;
  int p = t & 15;
  int py = p >> 2, px = p & 3;
  const float* ib = in + (size_t)img * Cin * Sin * Sin;
  float mx = -1e30f;
#pragma unroll
  for (int dy = 0; dy < 2; ++dy)
#pragma unroll
    for (int dx = 0; dx < 2; ++dx) {
      int oy = 2 * py + dy, ox = 2 * px + dx;
      float v = bias[co];
      for (int ci = 0; ci < Cin; ++ci) {
        const float* ip = ib + ((size_t)ci * Sin + oy) * Sin + ox;
        const float* wp = w + ((size_t)co * Cin + ci) * 9;
        v += ip[0] * wp[0] + ip[1] * wp[1] + ip[2] * wp[2]
           + ip[Sin] * wp[3] + ip[Sin + 1] * wp[4] + ip[Sin + 2] * wp[5]
           + ip[2 * Sin] * wp[6] + ip[2 * Sin + 1] * wp[7] + ip[2 * Sin + 2] * wp[8];
      }
      mx = fmaxf(mx, v);
    }
  float r = fmaxf(mx, 0.f);
  Hf[(size_t)img * 160 + t] = r;
  Hfh[(size_t)img * 160 + t] = (_Float16)r;
  HfT[((size_t)(img >> 7) * 160 + t) * 128 + (img & 127)] = (_Float16)r;
}

__global__ void f32_to_f16(const float* __restrict__ x, _Float16* __restrict__ y, int n) {
  int i = blockIdx.x * blockDim.x + threadIdx.x;
  if (i < n) y[i] = (_Float16)x[i];
}

// ---------------------------------------------------------------------------
// C = tanh(A @ W^T + bias). A f16 [M x K], W f16 [N x K]. One wave per 16x16
// tile, block=256 (8 waves), grid.x = ceil(tiles/8).
// Writes f16 row-major; f32 row-major if outf; per-batch transposed f16
// (layout [M/128][N][128]) if outT.
// ---------------------------------------------------------------------------
__global__ void gemm_aw_tanh(const _Float16* __restrict__ A,
                             const _Float16* __restrict__ W,
                             const float* __restrict__ bias,
                             _Float16* __restrict__ outh, float* __restrict__ outf,
                             _Float16* __restrict__ outT,
                             int M, int N, int K) {
  int lane = threadIdx.x & 31;
  int wave = threadIdx.x >> 5;
  int Mt = M >> 4, Nt = N >> 4;
  int tile = blockIdx.x * 8 + wave;
  if (tile >= Mt * Nt) return;             // wave-uniform: EXEC stays all-ones
  int mt = tile / Nt, nt = tile % Nt;
  const _Float16* Ab = A + (size_t)(mt * 16) * K;
  const _Float16* Wb = W + (size_t)(nt * 16) * K;
  v8f c = {};
  for (int k0 = 0; k0 < K; k0 += 32) {
    v16h a = load_fragA(Ab + k0, K, lane);
    v16h b = load_fragB_W(Wb + k0, K, lane);
    c = wmma_f16(a, b, c);
  }
  int n = nt * 16 + (lane & 15);
  int mbase = mt * 16 + ((lane >> 4) << 3);
  float bn = bias[n];
#pragma unroll
  for (int r = 0; r < 8; ++r) {
    int m = mbase + r;
    float v = tanhf(c[r] + bn);
    outh[(size_t)m * N + n] = (_Float16)v;
    if (outf) outf[(size_t)m * N + n] = v;
    if (outT) outT[((size_t)(m >> 7) * N + n) * 128 + (m & 127)] = (_Float16)v;
  }
}

// G[b] = X[b] @ X[b]^T, X f16 [8][128][512], G f32 [8][128][128].
// grid=(8 blocks x 8 batches), block=256: 64 waves/batch = 64 tiles.
__global__ void gram_kernel(const _Float16* __restrict__ X, float* __restrict__ G) {
  int b = blockIdx.y;
  const _Float16* Xb = X + (size_t)b * 128 * 512;
  float* Gb = G + (size_t)b * 128 * 128;
  int lane = threadIdx.x & 31;
  int wave = threadIdx.x >> 5;
  int tile = blockIdx.x * 8 + wave;        // 0..63
  int mt = tile >> 3, nt = tile & 7;
  const _Float16* Ab = Xb + (size_t)(mt * 16) * 512;
  const _Float16* Bb = Xb + (size_t)(nt * 16) * 512;  // B col n = X row n
  v8f c = {};
  for (int k0 = 0; k0 < 512; k0 += 32) {
    v16h a = load_fragA(Ab + k0, 512, lane);
    v16h bb = load_fragB_W(Bb + k0, 512, lane);
    c = wmma_f16(a, bb, c);
  }
  int n = nt * 16 + (lane & 15);
  int mbase = mt * 16 + ((lane >> 4) << 3);
#pragma unroll
  for (int r = 0; r < 8; ++r) Gb[(size_t)(mbase + r) * 128 + n] = c[r];
}

// Row softmax of (2G_ij - G_ii - G_jj)/sigma (mean-centering is softmax-
// invariant, so it is dropped). grid=(128 rows, 8 batches), block=128.
__global__ void softmax_kernel(const float* __restrict__ G, const float* __restrict__ sigma,
                               _Float16* __restrict__ Ksm) {
  int b = blockIdx.y, i = blockIdx.x, j = threadIdx.x;
  const float* Gb = G + (size_t)b * 128 * 128;
  float inv_sig = 1.f / sigma[0];
  float s = (2.f * Gb[(size_t)i * 128 + j] - Gb[(size_t)i * 128 + i]
             - Gb[(size_t)j * 128 + j]) * inv_sig;
  __shared__ float red[128];
  red[j] = s;
  __syncthreads();
  for (int off = 64; off; off >>= 1) {
    if (j < off) red[j] = fmaxf(red[j], red[j + off]);
    __syncthreads();
  }
  float mx = red[0];
  __syncthreads();
  float e = expf(s - mx);
  red[j] = e;
  __syncthreads();
  for (int off = 64; off; off >>= 1) {
    if (j < off) red[j] += red[j + off];
    __syncthreads();
  }
  Ksm[((size_t)b * 128 + i) * 128 + j] = (_Float16)(e / red[0]);
}

// A_out = 0.5 * (Z + Ksm @ Z) per batch. Ksm f16 [8][128][128] row-major,
// Z supplied TRANSPOSED: Zt f16 [8][160][128] (Zt[b][n][k] = Z[b][k][n]) so the
// B fragment is 2x 16B contiguous loads per lane. Output row-major [8][128][160].
// grid=(10 blocks x 8 batches), block=256: 80 waves/batch = 8x10 tiles.
__global__ void mp_kernel(const _Float16* __restrict__ Ksm, const _Float16* __restrict__ Zt,
                          _Float16* __restrict__ Aout) {
  int b = blockIdx.y;
  const _Float16* Kb = Ksm + (size_t)b * 128 * 128;
  const _Float16* Ztb = Zt + (size_t)b * 160 * 128;
  _Float16* Ab = Aout + (size_t)b * 128 * 160;
  int lane = threadIdx.x & 31;
  int wave = threadIdx.x >> 5;
  int tile = blockIdx.x * 8 + wave;        // 0..79
  if (tile >= 80) return;
  int mt = tile / 10, nt = tile % 10;
  v8f c = {};
  for (int k0 = 0; k0 < 128; k0 += 32) {
    v16h a = load_fragA(Kb + (size_t)(mt * 16) * 128 + k0, 128, lane);
    v16h bb = load_fragB_W(Ztb + (size_t)(nt * 16) * 128 + k0, 128, lane);
    c = wmma_f16(a, bb, c);
  }
  int n = nt * 16 + (lane & 15);
  int mbase = mt * 16 + ((lane >> 4) << 3);
#pragma unroll
  for (int r = 0; r < 8; ++r) {
    int m = mbase + r;
    float z = (float)Ztb[(size_t)n * 128 + m];   // Z[m][n]
    Ab[(size_t)m * 160 + n] = (_Float16)(0.5f * (c[r] + z));
  }
}

// out[b] = exp( sum_f max_n(H6[b][n][f]) * Wc[f] + bc ). grid=8, block=160.
__global__ void final_kernel(const float* __restrict__ H6, const float* __restrict__ Wc,
                             const float* __restrict__ bc, float* __restrict__ out) {
  int b = blockIdx.x;
  int f = threadIdx.x;  // 0..159
  const float* Hb = H6 + (size_t)b * 128 * 160;
  float mx = -1e30f;
  for (int n = 0; n < 128; ++n) mx = fmaxf(mx, Hb[(size_t)n * 160 + f]);
  __shared__ float acc[160];
  acc[f] = mx * Wc[f];
  __syncthreads();
  if (f == 0) {
    float s = 0.f;
    for (int i = 0; i < 160; ++i) s += acc[i];
    out[b] = expf(s + bc[0]);
  }
}

// ---------------------------------------------------------------------------
extern "C" void kernel_launch(void* const* d_in, const int* in_sizes, int n_in,
                              void* d_out, int out_size, void* d_ws, size_t ws_size,
                              hipStream_t stream) {
  (void)in_sizes; (void)n_in; (void)out_size; (void)ws_size;
  const float* H    = (const float*)d_in[0];
  const float* w1   = (const float*)d_in[1];
  const float* b1   = (const float*)d_in[2];
  const float* w2   = (const float*)d_in[3];
  const float* b2   = (const float*)d_in[4];
  const float* w3   = (const float*)d_in[5];
  const float* b3   = (const float*)d_in[6];
  const float* w4   = (const float*)d_in[7];
  const float* b4   = (const float*)d_in[8];
  const float* Wdk1 = (const float*)d_in[9];
  const float* bdk1 = (const float*)d_in[10];
  const float* Wdk2 = (const float*)d_in[11];
  const float* bdk2 = (const float*)d_in[12];
  const float* Wt1  = (const float*)d_in[13];
  const float* bt1  = (const float*)d_in[14];
  const float* Wt2  = (const float*)d_in[15];
  const float* bt2  = (const float*)d_in[16];
  const float* Wt3  = (const float*)d_in[17];
  const float* bt3  = (const float*)d_in[18];
  const float* sig  = (const float*)d_in[19];
  const float* Wc   = (const float*)d_in[20];
  const float* bc   = (const float*)d_in[21];
  float* out = (float*)d_out;

  // workspace bump allocator (256B aligned)
  char* base = (char*)d_ws;
  size_t off = 0;
  auto alloc = [&](size_t bytes) -> void* {
    off = (off + 255) & ~((size_t)255);
    void* p = (void*)(base + off);
    off += bytes;
    return p;
  };
  const int IMG = 1024;  // B*N
  float*     C1    = (float*)alloc((size_t)IMG * 10 * 49 * 49 * 4);
  float*     C2    = (float*)alloc((size_t)IMG * 10 * 23 * 23 * 4);
  float*     C3    = (float*)alloc((size_t)IMG * 10 * 10 * 10 * 4);
  float*     Hf    = (float*)alloc((size_t)IMG * 160 * 4);
  _Float16*  Hfh   = (_Float16*)alloc((size_t)IMG * 160 * 2);
  _Float16*  HfT   = (_Float16*)alloc((size_t)IMG * 160 * 2);   // [8][160][128]
  _Float16*  t1h   = (_Float16*)alloc((size_t)IMG * 256 * 2);
  _Float16*  dkfh  = (_Float16*)alloc((size_t)IMG * 512 * 2);
  float*     G     = (float*)alloc((size_t)8 * 128 * 128 * 4);
  _Float16*  Ksm   = (_Float16*)alloc((size_t)8 * 128 * 128 * 2);
  _Float16*  Ah    = (_Float16*)alloc((size_t)IMG * 160 * 2);
  _Float16*  Hcur  = (_Float16*)alloc((size_t)IMG * 160 * 2);
  _Float16*  HcurT = (_Float16*)alloc((size_t)IMG * 160 * 2);   // [8][160][128]
  float*     H6f   = (float*)alloc((size_t)IMG * 160 * 4);
  _Float16*  Wdk1h = (_Float16*)alloc((size_t)256 * 160 * 2);
  _Float16*  Wdk2h = (_Float16*)alloc((size_t)512 * 256 * 2);
  _Float16*  Wt1h  = (_Float16*)alloc((size_t)160 * 160 * 2);
  _Float16*  Wt2h  = (_Float16*)alloc((size_t)160 * 160 * 2);
  _Float16*  Wt3h  = (_Float16*)alloc((size_t)160 * 160 * 2);

  // weight conversions (f32 -> f16)
  auto cvt = [&](const float* src, _Float16* dst, int n) {
    f32_to_f16<<<(n + 255) / 256, 256, 0, stream>>>(src, dst, n);
  };
  cvt(Wdk1, Wdk1h, 256 * 160);
  cvt(Wdk2, Wdk2h, 512 * 256);
  cvt(Wt1, Wt1h, 160 * 160);
  cvt(Wt2, Wt2h, 160 * 160);
  cvt(Wt3, Wt3h, 160 * 160);

  // conv tower (fp32, memory-bound)
  conv_relu_pool<<<dim3(IMG, 10), 256, 1 * 9 * 4, stream>>>(H, w1, b1, C1, 1, 100);
  conv_relu_pool<<<dim3(IMG, 10), 256, 10 * 9 * 4, stream>>>(C1, w2, b2, C2, 10, 49);
  conv_relu_pool<<<dim3(IMG, 10), 256, 10 * 9 * 4, stream>>>(C2, w3, b3, C3, 10, 23);
  conv4_pack<<<IMG, 160, 0, stream>>>(C3, w4, b4, Hf, Hfh, HfT);

  // deep-kernel MLP: [1024,160]->[1024,256]->[1024,512]  (WMMA)
  {
    int tiles = (IMG / 16) * (256 / 16);
    gemm_aw_tanh<<<(tiles + 7) / 8, 256, 0, stream>>>(Hfh, Wdk1h, bdk1, t1h, nullptr,
                                                      nullptr, IMG, 256, 160);
  }
  {
    int tiles = (IMG / 16) * (512 / 16);
    gemm_aw_tanh<<<(tiles + 7) / 8, 256, 0, stream>>>(t1h, Wdk2h, bdk2, dkfh, nullptr,
                                                      nullptr, IMG, 512, 256);
  }

  // attention kernel matrix (computed ONCE, reused for all 3 message passes)
  gram_kernel<<<dim3(8, 8), 256, 0, stream>>>(dkfh, G);
  softmax_kernel<<<dim3(128, 8), 128, 0, stream>>>(G, sig, Ksm);

  // stage 1: A = 0.5*(Hf + K@Hf); H2 = tanh(A@Wt1^T + bt1)  (+ transposed copy)
  const int tilesT = (IMG / 16) * (160 / 16);  // 640
  mp_kernel<<<dim3(10, 8), 256, 0, stream>>>(Ksm, HfT, Ah);
  gemm_aw_tanh<<<(tilesT + 7) / 8, 256, 0, stream>>>(Ah, Wt1h, bt1, Hcur, nullptr,
                                                     HcurT, IMG, 160, 160);
  // stage 2
  mp_kernel<<<dim3(10, 8), 256, 0, stream>>>(Ksm, HcurT, Ah);
  gemm_aw_tanh<<<(tilesT + 7) / 8, 256, 0, stream>>>(Ah, Wt2h, bt2, Hcur, nullptr,
                                                     HcurT, IMG, 160, 160);
  // stage 3 (also write f32 for the max-reduction head; no transpose needed)
  mp_kernel<<<dim3(10, 8), 256, 0, stream>>>(Ksm, HcurT, Ah);
  gemm_aw_tanh<<<(tilesT + 7) / 8, 256, 0, stream>>>(Ah, Wt3h, bt3, Hcur, H6f,
                                                     nullptr, IMG, 160, 160);

  // head: per-batch feature max, dot with Wc, exp
  final_kernel<<<8, 160, 0, stream>>>(H6f, Wc, bc, out);
}
